// SHA_3530463117487
// MI455X (gfx1250) — matrix operand order repaired and checked by reading, not compiled, not run on hardware
//
#include <hip/hip_runtime.h>

// ---------------- types ----------------
typedef __attribute__((ext_vector_type(16))) __bf16 v16bf;
typedef __attribute__((ext_vector_type(8)))  float  v8f;

union FragB { v16bf v; uint4 u[2]; };

#define ROWS   16384   // B*S
#define SEQ    4096
#define DMODEL 1024
#define DK     128
#define NSPLIT 4
#define NEG_INF (-__builtin_inff())

// qscale = (1/sqrt(128)) * log2(e)  -> scores land in exp2 domain
#define QSCALE 0.12753257912782795f

__device__ inline unsigned short f2bf_bits(float x) {
    union { float f; unsigned u; } c; c.f = x;
    unsigned r = c.u + 0x7FFFu + ((c.u >> 16) & 1u);  // RNE
    return (unsigned short)(r >> 16);
}
__device__ inline void store_bf16(__bf16* p, float x) {
    *reinterpret_cast<unsigned short*>(p) = f2bf_bits(x);
}

// ---------------- kernel 0: transpose weights to bf16 [128][1024] ----------------
__global__ __launch_bounds__(256) void wtrans_kernel(const float* __restrict__ Wq,
                                                     const float* __restrict__ Wk,
                                                     const float* __restrict__ Wv,
                                                     __bf16* __restrict__ Wt) {
    const int mat = blockIdx.y;
    const float* W = (mat == 0) ? Wq : (mat == 1) ? Wk : Wv;
    const int idx = blockIdx.x * 256 + threadIdx.x;   // 0..131071
    const int n = idx >> 10;          // 0..127
    const int k = idx & 1023;         // 0..1023
    store_bf16(Wt + (size_t)mat * (DK * DMODEL) + idx, W[k * DK + n]);
}

// ---------------- kernel 1: projections Q,K,V via bf16 WMMA ----------------
__global__ __launch_bounds__(256) void proj_kernel(const float* __restrict__ X,
                                                   const __bf16* __restrict__ Wt,
                                                   const float* __restrict__ bq,
                                                   const float* __restrict__ bk,
                                                   const float* __restrict__ bv,
                                                   __bf16* __restrict__ Qb,
                                                   __bf16* __restrict__ Kb,
                                                   __bf16* __restrict__ Vt) {
    __shared__ __attribute__((aligned(16))) __bf16 xs[16 * DMODEL];  // 32 KB

    const int row0 = blockIdx.x * 16;
    const int t    = threadIdx.x;

    // stage 16x1024 X tile (contiguous 16384 floats) as bf16 into LDS
    const float* Xt = X + (size_t)row0 * DMODEL;
    for (int i = t; i < 16 * DMODEL; i += 256) store_bf16(&xs[i], Xt[i]);
    __syncthreads();

    const int lane = t & 31;
    const int nlo  = lane & 15;
    const int half = lane >> 4;
    const int n0   = (t >> 5) * 16;   // wave -> 16 output columns

    const __bf16* arow = xs + (size_t)nlo * DMODEL;  // A row for this lane

    for (int mat = 0; mat < 3; ++mat) {
        const __bf16* Wp = Wt + (size_t)mat * (DK * DMODEL) + (size_t)(n0 + nlo) * DMODEL;
        v8f c = {};
        for (int kt = 0; kt < DMODEL / 32; ++kt) {
            const int kbase = kt * 32;
            FragB a, b;
            a.u[0] = *reinterpret_cast<const uint4*>(arow + kbase + half * 8);
            a.u[1] = *reinterpret_cast<const uint4*>(arow + kbase + 16 + half * 8);
            const __bf16* wp = Wp + kbase + half * 8;
            b.u[0] = *reinterpret_cast<const uint4*>(wp);
            b.u[1] = *reinterpret_cast<const uint4*>(wp + 16);
            c = __builtin_amdgcn_wmma_f32_16x16x32_bf16(false, a.v, false, b.v,
                                                        (short)0, c, false, false);
        }
        const float* bias = (mat == 0) ? bq : (mat == 1) ? bk : bv;
        const float bval  = bias[n0 + nlo];
        for (int r = 0; r < 8; ++r) {
            const int   m    = half * 8 + r;
            const float val  = c[r] + bval;
            const size_t row = (size_t)(row0 + m);
            if (mat == 0)      store_bf16(&Qb[row * DK + n0 + nlo], val * QSCALE);
            else if (mat == 1) store_bf16(&Kb[row * DK + n0 + nlo], val);
            else               store_bf16(&Vt[(size_t)(n0 + nlo) * ROWS + row], val);
        }
    }
}

// ---------------- kernel 2: split-K causal flash attention (partial) ----------------
// grid = (1024 q-tiles, NSPLIT key-splits), block = 32 (one wave).
// Each split covers key tiles kb = split*32, split*32+128, ... and emits an
// UNNORMALIZED accumulator + per-row running (m, l) for the LSE combine.
__global__ __launch_bounds__(32) void attn_partial_kernel(const __bf16* __restrict__ Qb,
                                                          const __bf16* __restrict__ Kb,
                                                          const __bf16* __restrict__ Vt,
                                                          float* __restrict__ Opart,
                                                          float* __restrict__ Mpart,
                                                          float* __restrict__ Lpart) {
    __shared__ __attribute__((aligned(16))) __bf16 Pl[16 * 32];  // P transpose bounce

    const int lane = threadIdx.x & 31;
    const int nlo  = lane & 15;
    const int half = lane >> 4;

    const int qt    = blockIdx.x;
    const int split = blockIdx.y;
    const int qg0   = qt * 16;               // global query row base
    const int bat   = qg0 >> 12;             // /4096
    const int s0    = qg0 & (SEQ - 1);       // query base within batch

    // resident Q fragments (16x128 bf16 -> 4 A-frags)
    v16bf qf[4];
    {
        const __bf16* qrow = Qb + (size_t)(qg0 + nlo) * DK;
        for (int d = 0; d < 4; ++d) {
            FragB f;
            f.u[0] = *reinterpret_cast<const uint4*>(qrow + d * 32 + half * 8);
            f.u[1] = *reinterpret_cast<const uint4*>(qrow + d * 32 + 16 + half * 8);
            qf[d] = f.v;
        }
    }

    v8f zero = {};
    v8f acc[8];
    for (int i = 0; i < 8; ++i) acc[i] = zero;
    float mrow[8], lrow[8];
    for (int r = 0; r < 8; ++r) { mrow[r] = NEG_INF; lrow[r] = 0.0f; }

    const __bf16* Kbase = Kb + ((size_t)bat << 12) * DK;  // batch K rows
    const __bf16* Vbase = Vt + ((size_t)bat << 12);       // + dv*ROWS per column

    for (int kb = split * 32; kb < s0 + 16; kb += NSPLIT * 32) {
        // ---- scores: S0 (keys kb..kb+15), S1 (keys kb+16..kb+31)
        v8f sc0 = zero, sc1 = zero;
        for (int d = 0; d < 4; ++d) {
            FragB k0, k1;
            const __bf16* kp0 = Kbase + (size_t)(kb + nlo) * DK + d * 32 + half * 8;
            const __bf16* kp1 = kp0 + 16 * DK;
            k0.u[0] = *reinterpret_cast<const uint4*>(kp0);
            k0.u[1] = *reinterpret_cast<const uint4*>(kp0 + 16);
            k1.u[0] = *reinterpret_cast<const uint4*>(kp1);
            k1.u[1] = *reinterpret_cast<const uint4*>(kp1 + 16);
            sc0 = __builtin_amdgcn_wmma_f32_16x16x32_bf16(false, qf[d], false, k0.v,
                                                          (short)0, sc0, false, false);
            sc1 = __builtin_amdgcn_wmma_f32_16x16x32_bf16(false, qf[d], false, k1.v,
                                                          (short)0, sc1, false, false);
        }

        // ---- causal mask (only diagonal-crossing tiles)
        if (kb + 31 > s0) {
            for (int r = 0; r < 8; ++r) {
                const int sq = s0 + half * 8 + r;
                if (kb + nlo > sq)      sc0[r] = NEG_INF;
                if (kb + 16 + nlo > sq) sc1[r] = NEG_INF;
            }
        }

        // ---- online softmax (exp2 domain; scale folded into Q)
        float p0[8], p1[8];
        for (int r = 0; r < 8; ++r) {
            float tmax = fmaxf(sc0[r], sc1[r]);
            for (int off = 8; off >= 1; off >>= 1)
                tmax = fmaxf(tmax, __shfl_xor(tmax, off, 32));
            const float mnew = fmaxf(mrow[r], tmax);
            const float corr = exp2f(mrow[r] - mnew);
            const float e0 = exp2f(sc0[r] - mnew);
            const float e1 = exp2f(sc1[r] - mnew);
            float psum = e0 + e1;
            for (int off = 8; off >= 1; off >>= 1)
                psum += __shfl_xor(psum, off, 32);
            lrow[r] = lrow[r] * corr + psum;
            mrow[r] = mnew;
            p0[r] = e0; p1[r] = e1;
            for (int nt = 0; nt < 8; ++nt) acc[nt][r] *= corr;
        }

        // ---- transpose P (C-layout -> A-layout) through LDS
        __syncthreads();
        for (int r = 0; r < 8; ++r) {
            const int m = half * 8 + r;
            store_bf16(&Pl[m * 32 + nlo],       p0[r]);
            store_bf16(&Pl[m * 32 + 16 + nlo],  p1[r]);
        }
        __syncthreads();
        FragB pf;
        {
            const __bf16* prow = Pl + (size_t)nlo * 32;
            pf.u[0] = *reinterpret_cast<const uint4*>(prow + half * 8);
            pf.u[1] = *reinterpret_cast<const uint4*>(prow + 16 + half * 8);
        }

        // ---- O += P * V  (V stored transposed: contiguous keys per dv column)
        for (int nt = 0; nt < 8; ++nt) {
            FragB vf;
            const __bf16* vp = Vbase + (size_t)(nt * 16 + nlo) * ROWS + kb + half * 8;
            vf.u[0] = *reinterpret_cast<const uint4*>(vp);
            vf.u[1] = *reinterpret_cast<const uint4*>(vp + 16);
            acc[nt] = __builtin_amdgcn_wmma_f32_16x16x32_bf16(false, pf.v, false, vf.v,
                                                              (short)0, acc[nt], false, false);
        }
    }

    // ---- emit unnormalized partial + per-row stats
    const size_t pbase = ((size_t)qt * NSPLIT + split) * 16;
    for (int r = 0; r < 8; ++r) {
        const int m = half * 8 + r;
        float* orow = Opart + (pbase + m) * DK;
        for (int nt = 0; nt < 8; ++nt)
            orow[nt * 16 + nlo] = acc[nt][r];
    }
    if (nlo == 0) {
        for (int r = 0; r < 8; ++r) {
            const int m = half * 8 + r;
            Mpart[pbase + m] = mrow[r];
            Lpart[pbase + m] = lrow[r];
        }
    }
}

// ---------------- kernel 3: LSE combine of the NSPLIT partials ----------------
// grid = ROWS blocks, block = 128 threads (one per dv column).
__global__ __launch_bounds__(128) void attn_reduce_kernel(const float* __restrict__ Opart,
                                                          const float* __restrict__ Mpart,
                                                          const float* __restrict__ Lpart,
                                                          float* __restrict__ O) {
    const int row = blockIdx.x;            // 0..16383
    const int t   = threadIdx.x;           // dv column
    const int qt  = row >> 4;
    const int m   = row & 15;

    float ms[NSPLIT], ls[NSPLIT];
    float mm = NEG_INF;
    for (int s = 0; s < NSPLIT; ++s) {
        const size_t idx = ((size_t)qt * NSPLIT + s) * 16 + m;
        ms[s] = Mpart[idx];
        ls[s] = Lpart[idx];
        mm = fmaxf(mm, ms[s]);
    }
    float L = 0.0f, o = 0.0f;
    for (int s = 0; s < NSPLIT; ++s) {
        const float w = exp2f(ms[s] - mm);   // empty split: exp2(-inf)=0
        L += w * ls[s];
        o += w * Opart[(((size_t)qt * NSPLIT + s) * 16 + m) * DK + t];
    }
    O[(size_t)row * DK + t] = o / L;         // split 0 always non-empty => L > 0
}

// ---------------- launch ----------------
extern "C" void kernel_launch(void* const* d_in, const int* in_sizes, int n_in,
                              void* d_out, int out_size, void* d_ws, size_t ws_size,
                              hipStream_t stream) {
    const float* X  = (const float*)d_in[0];
    const float* Wq = (const float*)d_in[1];
    const float* bq = (const float*)d_in[2];
    const float* Wk = (const float*)d_in[3];
    const float* bk = (const float*)d_in[4];
    const float* Wv = (const float*)d_in[5];
    const float* bv = (const float*)d_in[6];
    float* O = (float*)d_out;

    // workspace layout (bytes):
    //  Wt 768KB | Qb 4MB | Kb 4MB | Vt 4MB | Opart 32MB | Mpart 256KB | Lpart 256KB
    char* ws = (char*)d_ws;
    __bf16* Wt    = (__bf16*)(ws);
    __bf16* Qb    = (__bf16*)(ws + 786432);
    __bf16* Kb    = (__bf16*)(ws + 786432 + 4194304);
    __bf16* Vt    = (__bf16*)(ws + 786432 + 2 * 4194304);
    float*  Opart = (float*) (ws + 786432 + 3 * 4194304);
    float*  Mpart = (float*) (ws + 786432 + 3 * 4194304 + 33554432);
    float*  Lpart = (float*) (ws + 786432 + 3 * 4194304 + 33554432 + 262144);

    wtrans_kernel<<<dim3(512, 3), 256, 0, stream>>>(Wq, Wk, Wv, Wt);
    proj_kernel<<<dim3(ROWS / 16), 256, 0, stream>>>(X, Wt, bq, bk, bv, Qb, Kb, Vt);
    attn_partial_kernel<<<dim3(ROWS / 16, NSPLIT), 32, 0, stream>>>(Qb, Kb, Vt,
                                                                    Opart, Mpart, Lpart);
    attn_reduce_kernel<<<dim3(ROWS), 128, 0, stream>>>(Opart, Mpart, Lpart, O);
}
